// DynamicConv_76948634075471
// MI455X (gfx1250) — matrix operand order repaired and checked by reading, not compiled, not run on hardware
//
#include <hip/hip_runtime.h>
#include <hip/hip_bf16.h>
#include <math.h>

// Dims from reference
#define BB   16
#define CC   256
#define OO   256
#define KK   4
#define KS   3
#define HH   64
#define WW   64
#define HW   4096        // 64*64

typedef __attribute__((ext_vector_type(16))) _Float16 v16h;
typedef __attribute__((ext_vector_type(8)))  _Float16 v8h;
typedef __attribute__((ext_vector_type(8)))  float    v8f;

union V16U { v16h v; v8h h[2]; };

// ---------------------------------------------------------------------------
// 1) Global average pool: x[B,C,H,W] -> pooled[B,C]
// ---------------------------------------------------------------------------
__global__ void dyn_pool_kernel(const float* __restrict__ x, float* __restrict__ pooled) {
    __shared__ float red[256];
    int bc  = blockIdx.x;            // 0..B*C-1
    int tid = threadIdx.x;
    const float* p = x + (size_t)bc * HW;
    float s = 0.f;
    for (int i = tid; i < HW; i += 256) s += p[i];
    red[tid] = s;
    __syncthreads();
    for (int off = 128; off > 0; off >>= 1) {
        if (tid < off) red[tid] += red[tid + off];
        __syncthreads();
    }
    if (tid == 0) pooled[bc] = red[0] * (1.f / (float)HW);
}

// ---------------------------------------------------------------------------
// 2) attn[b,k] = softmax_k( pooled[b,:] . attn_w[k,:] )   (tiny; one block)
// ---------------------------------------------------------------------------
__global__ void dyn_attn_kernel(const float* __restrict__ pooled,
                                const float* __restrict__ attn_w,
                                float* __restrict__ attn) {
    __shared__ float logits[BB * KK];
    int tid = threadIdx.x;
    if (tid < BB * KK) {
        int b = tid >> 2, k = tid & 3;
        float d = 0.f;
        for (int c = 0; c < CC; ++c)
            d += pooled[b * CC + c] * attn_w[k * CC + c];
        logits[tid] = d;
    }
    __syncthreads();
    if (tid < BB) {
        float m = -1e30f;
        for (int k = 0; k < KK; ++k) m = fmaxf(m, logits[tid * KK + k]);
        float e[KK], s = 0.f;
        for (int k = 0; k < KK; ++k) { e[k] = __expf(logits[tid * KK + k] - m); s += e[k]; }
        float inv = 1.f / s;
        for (int k = 0; k < KK; ++k) attn[tid * KK + k] = e[k] * inv;
    }
}

// ---------------------------------------------------------------------------
// 3) Combine expert kernels per sample, convert to f16 in layout
//    cw[b][kh][kw][o][c] = sum_k attn[b,k] * dyn[k,o,c,kh,kw]
//    Block = (b,o); thread = c.  Each thread reads its 9 contiguous taps per
//    expert (block covers 2304 contiguous floats per expert -> fully coalesced), writes
//    9 contiguous 256-half rows.
// ---------------------------------------------------------------------------
__global__ void dyn_combine_kernel(const float* __restrict__ dyn,
                                   const float* __restrict__ attn,
                                   _Float16* __restrict__ cw) {
    int b = blockIdx.x >> 8;         // grid.x = B*O = 4096
    int o = blockIdx.x & 255;
    int c = threadIdx.x;             // 256 threads
    float acc[9] = {};
#pragma unroll
    for (int k = 0; k < KK; ++k) {
        float a = attn[b * KK + k];
        const float* p = dyn + (((size_t)k * OO + o) * CC + c) * 9;
#pragma unroll
        for (int j = 0; j < 9; ++j) acc[j] += a * p[j];
    }
#pragma unroll
    for (int j = 0; j < 9; ++j)
        cw[(((size_t)(b * 9 + j)) * OO + o) * CC + c] = (_Float16)acc[j];
}

// ---------------------------------------------------------------------------
// 4) WMMA implicit-GEMM conv.  Per WG: 128 out-channels x 256 pixels (4 rows).
//    Grid: (H/4, O/128, B).  8 waves, each 32M x 128N = 16 wmma accumulators.
//    Dynamic LDS (99072 B): xt[6][66][32] f16 + wt[9][128][32] f16 so that a
//    channel-chunk needs only 2 barriers, then runs 144 WMMAs uninterrupted.
// ---------------------------------------------------------------------------
__global__ __launch_bounds__(256)
void dyn_conv_wmma_kernel(const float* __restrict__ x,
                          const _Float16* __restrict__ cw,
                          float* __restrict__ out) {
    extern __shared__ _Float16 smem[];
    _Float16* xt = smem;             // [row 0..5][w' 0..65][c 0..31]  (25344 B)
    _Float16* wt = smem + 6 * 66 * 32; // [tap 0..8][o 0..127][c 0..31] (73728 B)

    const int tid  = threadIdx.x;
    const int lane = tid & 31;
    const int wid  = tid >> 5;
    const int wm   = wid & 3;          // 4 M sub-tiles of 32
    const int wn   = wid >> 2;         // 2 N sub-tiles of 128
    const int mrow  = lane & 15;
    const int lhalf = lane >> 4;       // 0: lanes 0-15, 1: lanes 16-31
    const int alow  = lhalf * 8;       // A-frag K-interleave offset

    const int h0 = blockIdx.x * 4;
    const int o0 = blockIdx.y * 128;
    const int b  = blockIdx.z;

    v8f acc[2][8] = {};

    for (int cch = 0; cch < 8; ++cch) {           // channel chunks of 32
        const int c0 = cch * 32;
        __syncthreads();                           // prior compute done before LDS overwrite
        // ---- stage x tile (f32 -> f16, transposed, zero-padded halo) ----
        {
            const float* xb = x + (((size_t)b * CC + c0) << 12);
            for (int e = tid; e < 6 * 66 * 32; e += 256) {
                int c   = e / 396;
                int rem = e - c * 396;
                int r   = rem / 66;
                int wp  = rem - r * 66;
                int hg  = h0 + r - 1;
                int wg  = wp - 1;
                float v = 0.f;
                if ((unsigned)hg < 64u && (unsigned)wg < 64u)
                    v = xb[(c << 12) + (hg << 6) + wg];
                xt[(r * 66 + wp) * 32 + c] = (_Float16)v;
            }
        }
        // ---- stage all 9 weight taps [9][128 o][32 c] ----
        {
            const _Float16* wsrc = cw + (((size_t)b * 9 * OO + o0) << 8) + c0;
            for (int e = tid; e < 9 * 128 * 4; e += 256) {
                int chalf = e & 3;
                int o     = (e >> 2) & 127;
                int p     = e >> 9;
                // cw row index for (tap p, out-ch o0+o) = p*256 + o
                *(v8h*)&wt[(p * 128 + o) * 32 + chalf * 8] =
                    *(const v8h*)(wsrc + ((size_t)(p * 256 + o) << 8) + (chalf << 3));
            }
        }
        __syncthreads();

        // ---- prefetch next chunk (overlaps with the 144 WMMAs below) ----
        if (cch < 7) {
            const int c0n = c0 + 32;
            const float* xn = x + (((size_t)b * CC + c0n) << 12);
            for (int e = tid; e < 32 * 6 * 3; e += 256) {   // 128B segments of x rows
                int ci  = e / 18;
                int rr  = (e / 3) % 6;
                int seg = e % 3;
                int hg  = h0 + rr - 1;
                if ((unsigned)hg < 64u)
                    __builtin_prefetch(xn + ((size_t)ci << 12) + (hg << 6) + seg * 32, 0, 1);
            }
            const _Float16* wn_src = cw + c0n;
            for (int e = tid; e < 9 * 128; e += 256) {      // one 64B weight row each
                int j = e >> 7, o = e & 127;
                __builtin_prefetch(wn_src + (((size_t)(b * 9 + j) * OO + o0 + o) << 8), 0, 1);
            }
        }

        // ---- 9 taps x 16 WMMA, no barriers inside ----
#pragma unroll
        for (int p = 0; p < 9; ++p) {
            const int kh = p / 3, kw = p % 3;
            // A fragments (16x32 f16, interleaved K-half layout per ISA table)
            V16U af[2];
#pragma unroll
            for (int am = 0; am < 2; ++am) {
                int obase = (p * 128 + wm * 32 + am * 16 + mrow) * 32;
                af[am].h[0] = *(const v8h*)&wt[obase + alow];
                af[am].h[1] = *(const v8h*)&wt[obase + 16 + alow];
            }
#pragma unroll
            for (int bn = 0; bn < 8; ++bn) {
                int pix = wn * 128 + bn * 16 + mrow;
                int r   = (pix >> 6) + kh;
                int wp  = (pix & 63) + kw;
                int bbase = (r * 66 + wp) * 32 + lhalf * 8; // lanes16-31: K 8..15, 24..31
                V16U bf;
                bf.h[0] = *(const v8h*)&xt[bbase];
                bf.h[1] = *(const v8h*)&xt[bbase + 16];
                acc[0][bn] = __builtin_amdgcn_wmma_f32_16x16x32_f16(
                    false, af[0].v, false, bf.v, (short)0, acc[0][bn], false, false);
                acc[1][bn] = __builtin_amdgcn_wmma_f32_16x16x32_f16(
                    false, af[1].v, false, bf.v, (short)0, acc[1][bn], false, false);
            }
        }
    }

    // ---- epilogue: D layout lane l, vgpr r -> M = r + 8*(l>=16), N = l%16 ----
#pragma unroll
    for (int am = 0; am < 2; ++am) {
        int obase = o0 + wm * 32 + am * 16 + lhalf * 8;
#pragma unroll
        for (int bn = 0; bn < 8; ++bn) {
            int pix = wn * 128 + bn * 16 + mrow;
            int h = h0 + (pix >> 6), w = pix & 63;
#pragma unroll
            for (int r = 0; r < 8; ++r) {
                out[(((size_t)b * OO + (obase + r)) << 12) + (h << 6) + w] = acc[am][bn][r];
            }
        }
    }
}

// ---------------------------------------------------------------------------
// 5) BN batch stats per channel -> scale/bias
// ---------------------------------------------------------------------------
__global__ void dyn_bnstats_kernel(const float* __restrict__ out,
                                   const float* __restrict__ gamma,
                                   const float* __restrict__ beta,
                                   float* __restrict__ scale,
                                   float* __restrict__ bias) {
    __shared__ float r1[256];
    __shared__ float r2[256];
    int o = blockIdx.x, tid = threadIdx.x;
    float s1 = 0.f, s2 = 0.f;
    for (int b = 0; b < BB; ++b) {
        const float* p = out + (((size_t)b * OO + o) << 12);
        for (int i = tid; i < HW; i += 256) {
            float v = p[i];
            s1 += v; s2 += v * v;
        }
    }
    r1[tid] = s1; r2[tid] = s2;
    __syncthreads();
    for (int off = 128; off > 0; off >>= 1) {
        if (tid < off) { r1[tid] += r1[tid + off]; r2[tid] += r2[tid + off]; }
        __syncthreads();
    }
    if (tid == 0) {
        const float n = (float)(BB * HW);
        float mean = r1[0] / n;
        float var  = r2[0] / n - mean * mean;
        float sc   = gamma[o] * rsqrtf(var + 1e-3f);
        scale[o] = sc;
        bias[o]  = beta[o] - mean * sc;
    }
}

// ---------------------------------------------------------------------------
// 6) BN apply + SiLU (in place on d_out)
// ---------------------------------------------------------------------------
__global__ void dyn_bnsilu_kernel(float* __restrict__ out,
                                  const float* __restrict__ scale,
                                  const float* __restrict__ bias) {
    size_t idx = (size_t)blockIdx.x * 256 + threadIdx.x;  // < B*O*H*W
    int o = (int)((idx >> 12) & 255);
    float y = out[idx] * scale[o] + bias[o];
    out[idx] = y / (1.f + __expf(-y));
}

// ---------------------------------------------------------------------------
extern "C" void kernel_launch(void* const* d_in, const int* in_sizes, int n_in,
                              void* d_out, int out_size, void* d_ws, size_t ws_size,
                              hipStream_t stream) {
    const float* x      = (const float*)d_in[0];   // [16,256,64,64]
    const float* dyn    = (const float*)d_in[1];   // [4,256,256,3,3]
    const float* attn_w = (const float*)d_in[2];   // [4,256]
    const float* gamma  = (const float*)d_in[3];   // [256]
    const float* beta   = (const float*)d_in[4];   // [256]
    float* out = (float*)d_out;                    // [16,256,64,64]

    // workspace layout (needs ~18.9 MB)
    char* ws = (char*)d_ws;
    float*    pooled  = (float*)(ws);              // 16*256 f32
    float*    attn    = (float*)(ws + 16384);      // 16*4 f32
    float*    bnscale = (float*)(ws + 16640);      // 256 f32
    float*    bnbias  = (float*)(ws + 17664);      // 256 f32
    _Float16* cw      = (_Float16*)(ws + 32768);   // 16*9*256*256 f16

    // conv kernel uses 99072 B of dynamic LDS (gfx1250 WGP has 320 KB)
    static const int conv_lds = (6 * 66 * 32 + 9 * 128 * 32) * (int)sizeof(_Float16);
    hipFuncSetAttribute((const void*)dyn_conv_wmma_kernel,
                        hipFuncAttributeMaxDynamicSharedMemorySize, conv_lds);

    dyn_pool_kernel<<<BB * CC, 256, 0, stream>>>(x, pooled);
    dyn_attn_kernel<<<1, 256, 0, stream>>>(pooled, attn_w, attn);
    dyn_combine_kernel<<<BB * OO, 256, 0, stream>>>(dyn, attn, cw);
    dim3 gconv(HH / 4, OO / 128, BB);
    dyn_conv_wmma_kernel<<<gconv, 256, conv_lds, stream>>>(x, cw, out);
    dyn_bnstats_kernel<<<OO, 256, 0, stream>>>(out, gamma, beta, bnscale, bnbias);
    dyn_bnsilu_kernel<<<(BB * OO * HW) / 256, 256, 0, stream>>>(out, bnscale, bnbias);
}
